// MaskedLinear_57690000719893
// MI455X (gfx1250) — compile-verified, matrix operand order checked by reading, not run on hardware
//
#include <hip/hip_runtime.h>
#include <hip/hip_bf16.h>

typedef __attribute__((ext_vector_type(4)))  int    v4i;
typedef __attribute__((ext_vector_type(4)))  float  v4f;
typedef __attribute__((ext_vector_type(8)))  float  v8f;
typedef __attribute__((ext_vector_type(4)))  __bf16 v4bf;
typedef __attribute__((ext_vector_type(8)))  __bf16 v8bf;
typedef __attribute__((ext_vector_type(16))) __bf16 v16bf;

#define AS1 __attribute__((address_space(1)))
#define AS3 __attribute__((address_space(3)))

#define M_DIM 8192   // B*S
#define N_DIM 4096   // D_OUT
#define K_DIM 4096   // D_IN
#define KTILE 64     // two v_wmma_f32_16x16x32_bf16 k-chunks per LDS stage
#define LDSS  72     // padded LDS row stride (bf16): 36 dwords -> conflict-free b128
#define NKT   (K_DIM / KTILE)   // 64

// ---- CDNA5 async global->LDS copy (ASYNCcnt path), 16B per lane ----
__device__ __forceinline__ void async_copy_b128(const __bf16* g, __bf16* l) {
#if __has_builtin(__builtin_amdgcn_global_load_async_to_lds_b128)
  __builtin_amdgcn_global_load_async_to_lds_b128(
      (AS1 v4i*)g, (AS3 v4i*)l, 0, 0);
#else
  unsigned lds_off = (unsigned)(unsigned long long)(AS3 void*)l;
  asm volatile("global_load_async_to_lds_b128 %0, %1, off"
               :: "v"(lds_off), "v"((unsigned long long)g)
               : "memory");
#endif
}

__device__ __forceinline__ void wait_async0() {
#if __has_builtin(__builtin_amdgcn_s_wait_asynccnt)
  __builtin_amdgcn_s_wait_asynccnt(0);
#else
  asm volatile("s_wait_asynccnt 0x0" ::: "memory");
#endif
}

// ---------------- prepass 1: wb = bf16(weight * mask) ----------------
__global__ __launch_bounds__(256) void masked_weight_bf16_kernel(
    const float* __restrict__ w, const float* __restrict__ m,
    __bf16* __restrict__ wb, int n4) {
  int i = blockIdx.x * blockDim.x + threadIdx.x;
  if (i >= n4) return;
  size_t off = (size_t)i * 4;
  v4f wv = *(const v4f*)(w + off);
  v4f mv = *(const v4f*)(m + off);
  v4bf o;
  o[0] = (__bf16)(wv[0] * mv[0]);
  o[1] = (__bf16)(wv[1] * mv[1]);
  o[2] = (__bf16)(wv[2] * mv[2]);
  o[3] = (__bf16)(wv[3] * mv[3]);
  *(v4bf*)(wb + off) = o;
}

// ---------------- prepass 2: xb = bf16(x) ----------------
__global__ __launch_bounds__(256) void x_bf16_kernel(
    const float* __restrict__ x, __bf16* __restrict__ xb, int n4) {
  int i = blockIdx.x * blockDim.x + threadIdx.x;
  if (i >= n4) return;
  size_t off = (size_t)i * 4;
  v4f xv = *(const v4f*)(x + off);
  v4bf o;
  o[0] = (__bf16)xv[0];
  o[1] = (__bf16)xv[1];
  o[2] = (__bf16)xv[2];
  o[3] = (__bf16)xv[3];
  *(v4bf*)(xb + off) = o;
}

// ---------------- WMMA GEMM: y = xb @ wb^T + bias ----------------
// Block tile 128(M) x 128(N), 8 wave32; wave tile 32(M) x 64(N) = 2x4 accs.
__global__ __launch_bounds__(256) void masked_linear_wmma_kernel(
    const __bf16* __restrict__ xb, const __bf16* __restrict__ wb,
    const float* __restrict__ bias, float* __restrict__ y) {
  __shared__ __align__(16) __bf16 ldsA[2][128 * LDSS];
  __shared__ __align__(16) __bf16 ldsB[2][128 * LDSS];

  const int t      = threadIdx.x;
  const int wave   = t >> 5;
  const int lane   = t & 31;
  const int laneLo = lane & 15;
  const int laneHi = lane >> 4;
  const int wm     = wave & 3;   // 4 waves along M (4*32 = 128)
  const int wn     = wave >> 2;  // 2 waves along N (2*64 = 128)
  const int bm     = blockIdx.x; // M/128 = 64
  const int bn     = blockIdx.y; // N/128 = 32

  // async staging: thread owns 32 contiguous k of one row (4x b128)
  const int srow = t >> 1;        // 0..127
  const int skb  = (t & 1) << 5;  // 0 or 32 elems

  const __bf16* xg = xb + (size_t)(bm * 128 + srow) * K_DIM + skb;
  const __bf16* wg = wb + (size_t)(bn * 128 + srow) * K_DIM + skb;

  auto stage = [&](int kt, int buf) {
    const __bf16* xp = xg + kt * KTILE;
    const __bf16* wp = wg + kt * KTILE;
    __bf16* ap = &ldsA[buf][srow * LDSS + skb];
    __bf16* bp = &ldsB[buf][srow * LDSS + skb];
#pragma unroll
    for (int c = 0; c < 4; ++c) {
      async_copy_b128(xp + c * 8, ap + c * 8);
      async_copy_b128(wp + c * 8, bp + c * 8);
    }
  };

  v8f acc[2][4] = {};

  // ISA 16-bit A-fragment swizzle: lanes 0-15 take K 0-7 (v0-3) & 16-23 (v4-7),
  // lanes 16-31 take K 8-15 & 24-31.  B = W^T, so W rows use the same swizzle.
  // Fragments loaded just-in-time: only av[2] (16 VGPRs) + one bv (8 VGPRs)
  // live at once -> allocator can keep acc=wmma(...,acc) fully in place.
  auto compute = [&](int buf) {
    const __bf16* As = &ldsA[buf][wm * 32 * LDSS];
    const __bf16* Bs = &ldsB[buf][wn * 64 * LDSS];
#pragma unroll
    for (int kc = 0; kc < 2; ++kc) {
      const int fo = laneLo * LDSS + kc * 32 + laneHi * 8;
      v16bf av[2];
#pragma unroll
      for (int tm = 0; tm < 2; ++tm) {
        v8bf a0 = *(const v8bf*)(As + tm * 16 * LDSS + fo);
        v8bf a1 = *(const v8bf*)(As + tm * 16 * LDSS + fo + 16);
        av[tm] = __builtin_shufflevector(a0, a1, 0, 1, 2, 3, 4, 5, 6, 7,
                                         8, 9, 10, 11, 12, 13, 14, 15);
      }
#pragma unroll
      for (int tn = 0; tn < 4; ++tn) {
        v8bf b0 = *(const v8bf*)(Bs + tn * 16 * LDSS + fo);
        v8bf b1 = *(const v8bf*)(Bs + tn * 16 * LDSS + fo + 16);
        v16bf bv = __builtin_shufflevector(b0, b1, 0, 1, 2, 3, 4, 5, 6, 7,
                                           8, 9, 10, 11, 12, 13, 14, 15);
#pragma unroll
        for (int tm = 0; tm < 2; ++tm)
          acc[tm][tn] = __builtin_amdgcn_wmma_f32_16x16x32_bf16(
              false, av[tm], false, bv, (short)0, acc[tm][tn], false, false);
      }
    }
  };

  // pipeline: stage tile 0
  stage(0, 0);
  wait_async0();
  __syncthreads();

#pragma unroll 1  // keep rolled: avoids cross-iteration live-range blowup
  for (int kt = 0; kt < NKT; ++kt) {
    const int  cur      = kt & 1;
    const bool has_next = (kt + 1) < NKT;
    if (has_next) stage(kt + 1, cur ^ 1);  // DMA overlaps WMMA below
    compute(cur);
    if (has_next) {
      wait_async0();     // my async copies landed in LDS (ASYNCcnt not in barrier)
      __syncthreads();
    }
  }

  // epilogue: C/D layout -> VGPR r holds M=r (lanes 0-15) / M=r+8 (lanes 16-31)
  const int n_base = bn * 128 + wn * 64;
  const int m_base = bm * 128 + wm * 32;
#pragma unroll
  for (int tn = 0; tn < 4; ++tn) {
    const int   n   = n_base + tn * 16 + laneLo;
    const float bvv = bias[n];
#pragma unroll
    for (int tm = 0; tm < 2; ++tm) {
      const int m0 = m_base + tm * 16 + laneHi * 8;
#pragma unroll
      for (int r = 0; r < 8; ++r) {
        y[(size_t)(m0 + r) * N_DIM + n] = acc[tm][tn][r] + bvv;
      }
    }
  }
}

extern "C" void kernel_launch(void* const* d_in, const int* in_sizes, int n_in,
                              void* d_out, int out_size, void* d_ws, size_t ws_size,
                              hipStream_t stream) {
  const float* x    = (const float*)d_in[0];
  const float* w    = (const float*)d_in[1];
  const float* bias = (const float*)d_in[2];
  const float* mask = (const float*)d_in[3];

  // workspace layout: [wb: 32 MiB][xb: 64 MiB]  (needs 96 MiB scratch)
  __bf16* wb = (__bf16*)d_ws;
  __bf16* xbv = (__bf16*)((char*)d_ws + (size_t)N_DIM * K_DIM * sizeof(__bf16));

  const int nw4 = (N_DIM * K_DIM) / 4;  // 16384 blocks
  masked_weight_bf16_kernel<<<nw4 / 256, 256, 0, stream>>>(w, mask, wb, nw4);

  const int nx4 = (M_DIM * K_DIM) / 4;  // 32768 blocks
  x_bf16_kernel<<<nx4 / 256, 256, 0, stream>>>(x, xbv, nx4);

  dim3 grid(M_DIM / 128, N_DIM / 128);  // 64 x 32 = 2048 workgroups
  masked_linear_wmma_kernel<<<grid, 256, 0, stream>>>(xbv, wb, bias,
                                                      (float*)d_out);
}
